// HybridSemanticFusion_19095424598634
// MI455X (gfx1250) — matrix operand order repaired and checked by reading, not compiled, not run on hardware
//
#include <hip/hip_runtime.h>
#include <stdint.h>

// Problem constants (from reference): L=4, B=16, N=4096, D=1024, k=20, k_sel=100
#define LNUM 4
#define BNUM 16
#define NTOK 4096
#define DDIM 1024
#define KCLU 20
#define KSEL 100
#define LD   (LNUM * DDIM)   // 4096 concat feature dim
#define MPAD 112             // 7 * 16 point tiles
#define KCP  32              // 2 * 16 cluster tiles
#define KM_ITERS 10

typedef __attribute__((ext_vector_type(16))) __bf16 v16bf;
typedef __attribute__((ext_vector_type(8)))  float  v8f;

union BfFrag { v16bf v; uint32_t u[8]; };

__device__ __forceinline__ unsigned short f2bf(float f) {
  union { float f; uint32_t u; } x; x.f = f;
  uint32_t u = x.u;
  uint32_t r = u + 0x7FFFu + ((u >> 16) & 1u);   // round-to-nearest-even
  return (unsigned short)(r >> 16);
}

// ---------------- Kernel 1: anomaly score = softmax(mean_L)[...,1] ----------
__global__ void hsf_score(const float* __restrict__ amaps, float* __restrict__ score) {
  int i = blockIdx.x * blockDim.x + threadIdx.x;       // i over B*N
  if (i >= BNUM * NTOK) return;
  float m0 = 0.f, m1 = 0.f;
  #pragma unroll
  for (int l = 0; l < LNUM; ++l) {
    int base = ((l * BNUM * NTOK) + i) * 2;
    m0 += amaps[base + 0];
    m1 += amaps[base + 1];
  }
  m0 *= (1.f / LNUM); m1 *= (1.f / LNUM);
  score[i] = 1.f / (1.f + __expf(m0 - m1));            // softmax class-1 prob
}

// ---------------- Kernel 2: per-batch top-100 (stable argmax rounds) --------
__global__ void hsf_topk(const float* __restrict__ score, int* __restrict__ topidx) {
  __shared__ float sv[NTOK];
  __shared__ float rv[256];
  __shared__ int   ri[256];
  const int b = blockIdx.x, t = threadIdx.x;
  for (int i = t; i < NTOK; i += 256) sv[i] = score[b * NTOK + i];
  __syncthreads();
  for (int r = 0; r < KSEL; ++r) {
    float bv = -3.0e38f; int bi = -1;
    for (int i = t; i < NTOK; i += 256) {
      float v = sv[i];
      if (v > bv || (v == bv && i < bi)) { bv = v; bi = i; }
    }
    rv[t] = bv; ri[t] = bi; __syncthreads();
    for (int off = 128; off > 0; off >>= 1) {
      if (t < off) {
        if (rv[t + off] > rv[t] || (rv[t + off] == rv[t] && ri[t + off] < ri[t])) {
          rv[t] = rv[t + off]; ri[t] = ri[t + off];
        }
      }
      __syncthreads();
    }
    if (t == 0) { topidx[b * KSEL + r] = ri[0]; sv[ri[0]] = -3.0e38f; }
    __syncthreads();
  }
}

// ---------------- Kernel 3: gather top tokens -> fp32 stacked + bf16 copy ---
__global__ void hsf_gather(const float* __restrict__ patch,
                           const int* __restrict__ topidx,
                           float* __restrict__ xf,
                           unsigned short* __restrict__ xbf) {
  int i = blockIdx.x * blockDim.x + threadIdx.x;       // over B*MPAD*LD
  if (i >= BNUM * MPAD * LD) return;
  int b = i / (MPAD * LD);
  int rem = i - b * (MPAD * LD);
  int p = rem / LD;
  int f = rem - p * LD;                                // layer-major feature index
  if (p < KSEL) {
    int l = f >> 10, d = f & (DDIM - 1);
    int idx = topidx[b * KSEL + p];
    float v = patch[(((l * BNUM + b) * NTOK + idx) * DDIM) + d];
    xf[(b * KSEL + p) * LD + f] = v;
    xbf[(b * MPAD + p) * LD + f] = f2bf(v);
  } else {
    xbf[(b * MPAD + p) * LD + f] = 0;                  // zero pad rows 100..111
  }
}

// ---------------- Kernel 4: layer-averaged tokens ---------------------------
__global__ void hsf_avg(const float* __restrict__ xf, float* __restrict__ avg) {
  int i = blockIdx.x * blockDim.x + threadIdx.x;       // over B*KSEL*DDIM
  if (i >= BNUM * KSEL * DDIM) return;
  int bp = i / DDIM, d = i - bp * DDIM;
  const float* row = xf + bp * LD;
  float s = 0.f;
  #pragma unroll
  for (int l = 0; l < LNUM; ++l) s += row[l * DDIM + d];
  avg[i] = s * (1.f / LNUM);
}

// ---------------- Kernel 5: persistent per-batch k-means (WMMA assign) ------
__device__ __forceinline__ float block_reduce_128(float v, float* s_red, int t) {
  s_red[t] = v; __syncthreads();
  for (int off = 64; off > 0; off >>= 1) {
    if (t < off) s_red[t] += s_red[t + off];
    __syncthreads();
  }
  float r = s_red[0]; __syncthreads();
  return r;
}

__global__ void __launch_bounds__(128)
hsf_kmeans(const float* __restrict__ xf_all,
           const unsigned short* __restrict__ xbf_all,
           const float* __restrict__ avg_all,
           float* __restrict__ cf_all,
           unsigned short* __restrict__ cbf_all,
           float* __restrict__ out) {
  __shared__ float s_acc[KCLU * 512];   // centroid-update slice accumulators
  __shared__ float s_G[MPAD * KCP];     // x . c^T
  __shared__ int   s_lab[MPAD];
  __shared__ float s_csq[KCP];
  __shared__ int   s_cnt[KCP];
  __shared__ float s_red[128];

  const int b = blockIdx.x, t = threadIdx.x;
  const float*          xf  = xf_all  + (size_t)b * KSEL * LD;
  const unsigned short* xbf = xbf_all + (size_t)b * MPAD * LD;
  const float*          avg = avg_all + (size_t)b * KSEL * DDIM;
  float*                cf  = cf_all  + (size_t)b * KCLU * LD;
  unsigned short*       cbf = cbf_all + (size_t)b * KCP  * LD;

  // ---- init: c0 = x[linspace(0,99,20).astype(int)] ----
  for (int e = t; e < KCLU * LD; e += 128) {
    int k = e / LD, d = e - k * LD;
    int src = (k * (KSEL - 1)) / (KCLU - 1);           // floor(k*99/19)
    float v = xf[src * LD + d];
    cf[e] = v;
    cbf[k * LD + d] = f2bf(v);
  }
  for (int e = t; e < (KCP - KCLU) * LD; e += 128) cbf[KCLU * LD + e] = 0;
  __syncthreads();
  // |c|^2 for the initial centroids
  for (int k = 0; k < KCLU; ++k) {
    float part = 0.f;
    for (int d = t; d < LD; d += 128) { float v = cf[k * LD + d]; part += v * v; }
    float s = block_reduce_128(part, s_red, t);
    if (t == 0) s_csq[k] = s;
  }
  if (t >= KCLU && t < KCP) s_csq[t] = 3.0e38f;        // padded clusters never win
  __syncthreads();

  const int wv = t >> 5, lane = t & 31;
  const int r = lane & 15, half = lane >> 4;

  for (int it = 0; it <= KM_ITERS; ++it) {
    // ===== assign: G = x_bf16 . c_bf16^T via v_wmma_f32_16x16x32_bf16 =====
    for (int tile = wv; tile < 14; tile += 4) {        // 7 M-tiles x 2 N-tiles
      const int mt = tile >> 1, nt = tile & 1;
      v8f acc = {0.f, 0.f, 0.f, 0.f, 0.f, 0.f, 0.f, 0.f};
      const unsigned short* arow = xbf + (mt * 16 + r) * LD;   // point row (M=r)
      const unsigned short* brow = cbf + (nt * 16 + r) * LD;   // cluster row (N=r)
      for (int kk = 0; kk < LD; kk += 32) {
        BfFrag a, bb;
        #pragma unroll
        for (int j = 0; j < 8; ++j) {
          // A 16x32 bf16 lane layout: lanes<16 K={0..7,16..23}, lanes>=16 +8
          int ka = ((j >> 2) << 4) + (half << 3) + ((j & 3) << 1);
          a.u[j] = *(const uint32_t*)(arow + kk + ka);
          // B 32x16 bf16 lane layout: lanes<16 K={0..7,16..23}, lanes>=16 +8
          int kb = ((j >> 2) << 4) + (half << 3) + ((j & 3) << 1);
          bb.u[j] = *(const uint32_t*)(brow + kk + kb);
        }
        acc = __builtin_amdgcn_wmma_f32_16x16x32_bf16(
            false, a.v, false, bb.v, (short)0, acc, false, false);
      }
      #pragma unroll
      for (int v = 0; v < 8; ++v)                      // D: M = v + 8*half
        s_G[(mt * 16 + v + (half << 3)) * KCP + nt * 16 + r] = acc[v];
    }
    __syncthreads();
    // argmin_k (|c_k|^2 - 2 x.c_k)   (|x|^2 constant per row)
    if (t < KSEL) {
      float best = s_csq[0] - 2.f * s_G[t * KCP + 0];
      int bi = 0;
      #pragma unroll 4
      for (int k = 1; k < KCLU; ++k) {
        float d2 = s_csq[k] - 2.f * s_G[t * KCP + k];
        if (d2 < best) { best = d2; bi = k; }
      }
      s_lab[t] = bi;
    }
    if (t < KCP) s_cnt[t] = 0;
    __syncthreads();
    if (t < KSEL) atomicAdd(&s_cnt[s_lab[t]], 1);      // integer: deterministic
    __syncthreads();

    if (it == KM_ITERS) break;                         // 11th assign = final labels

    // ===== update: new centroids (race-free; fixed dim ownership) =====
    float csq_part[KCLU];
    #pragma unroll
    for (int k = 0; k < KCLU; ++k) csq_part[k] = 0.f;
    for (int s = 0; s < LD / 512; ++s) {
      for (int i = t; i < KCLU * 512; i += 128) s_acc[i] = 0.f;
      __syncthreads();
      for (int p = 0; p < KSEL; ++p) {                 // sequential p => no races
        int base = s_lab[p] * 512;
        const float* xr = xf + p * LD + s * 512;
        #pragma unroll
        for (int j = 0; j < 4; ++j) {
          int dl = t + j * 128;
          s_acc[base + dl] += xr[dl];
        }
      }
      __syncthreads();
      for (int k = 0; k < KCLU; ++k) {
        float inv = s_cnt[k] > 0 ? 1.f / (float)s_cnt[k] : 0.f;
        #pragma unroll
        for (int j = 0; j < 4; ++j) {
          int dl = t + j * 128, d = s * 512 + dl;
          float nv = s_cnt[k] > 0 ? s_acc[k * 512 + dl] * inv : cf[k * LD + d];
          cf[k * LD + d] = nv;
          cbf[k * LD + d] = f2bf(nv);
          csq_part[k] += nv * nv;
        }
      }
      __syncthreads();
    }
    for (int k = 0; k < KCLU; ++k) {
      float s = block_reduce_128(csq_part[k], s_red, t);
      if (t == 0) s_csq[k] = s;
    }
    __syncthreads();
  }

  // ===== centers: out_d = (1/K) sum_p avg[p,d] / cnt[label_p]; normalize =====
  float* s_w   = s_G;                                  // reuse: 100 weights
  float* s_out = s_acc;                                // reuse: 1024 dims
  if (t < KSEL) s_w[t] = 1.f / (float)s_cnt[s_lab[t]];
  __syncthreads();
  float nrm_part = 0.f;
  #pragma unroll
  for (int j = 0; j < DDIM / 128; ++j) {
    int d = t + j * 128;
    float s = 0.f;
    for (int p = 0; p < KSEL; ++p) s += s_w[p] * avg[p * DDIM + d];
    s *= (1.f / KCLU);
    s_out[d] = s;
    nrm_part += s * s;
  }
  float nrm2 = block_reduce_128(nrm_part, s_red, t);
  float inv_n = 1.f / fmaxf(sqrtf(nrm2), 1e-12f);
  #pragma unroll
  for (int j = 0; j < DDIM / 128; ++j) {
    int d = t + j * 128;
    out[b * DDIM + d] = s_out[d] * inv_n;
  }
}

// ---------------- Host launch ----------------------------------------------
extern "C" void kernel_launch(void* const* d_in, const int* in_sizes, int n_in,
                              void* d_out, int out_size, void* d_ws, size_t ws_size,
                              hipStream_t stream) {
  (void)in_sizes; (void)n_in; (void)out_size; (void)ws_size;
  const float* patch = (const float*)d_in[0];          // [L,B,N,D] fp32
  const float* amaps = (const float*)d_in[1];          // [L,B,N,2] fp32
  // d_in[2] = k_clusters (=20), baked into compile-time constants.

  char* ws = (char*)d_ws;
  size_t off = 0;
  auto alloc = [&](size_t bytes) { size_t o = off; off += (bytes + 255) & ~size_t(255); return o; };
  float*          score  = (float*)(ws + alloc((size_t)BNUM * NTOK * 4));
  int*            topidx = (int*)  (ws + alloc((size_t)BNUM * KSEL * 4));
  float*          xf     = (float*)(ws + alloc((size_t)BNUM * KSEL * LD * 4));
  unsigned short* xbf    = (unsigned short*)(ws + alloc((size_t)BNUM * MPAD * LD * 2));
  float*          avg    = (float*)(ws + alloc((size_t)BNUM * KSEL * DDIM * 4));
  float*          cf     = (float*)(ws + alloc((size_t)BNUM * KCLU * LD * 4));
  unsigned short* cbf    = (unsigned short*)(ws + alloc((size_t)BNUM * KCP * LD * 2));

  hsf_score<<<(BNUM * NTOK + 255) / 256, 256, 0, stream>>>(amaps, score);
  hsf_topk<<<BNUM, 256, 0, stream>>>(score, topidx);
  {
    int total = BNUM * MPAD * LD;
    hsf_gather<<<(total + 255) / 256, 256, 0, stream>>>(patch, topidx, xf, xbf);
  }
  {
    int total = BNUM * KSEL * DDIM;
    hsf_avg<<<(total + 255) / 256, 256, 0, stream>>>(xf, avg);
  }
  hsf_kmeans<<<BNUM, 128, 0, stream>>>(xf, xbf, avg, cf, cbf, (float*)d_out);
}
